// GCN_88244398063622
// MI455X (gfx1250) — matrix operand order repaired and checked by reading, not compiled, not run on hardware
//
#include <hip/hip_runtime.h>

#define DHID 128
#define NFEAT 64

typedef __bf16 bf16_t;
typedef bf16_t v16bf __attribute__((ext_vector_type(16)));
typedef bf16_t v8bf  __attribute__((ext_vector_type(8)));
typedef float  v8f   __attribute__((ext_vector_type(8)));

// A layout (matches B): lane(g,m) holds K = kbase + g*16 + [0..15]
// (elems 0..15), all of row m.
static __device__ __forceinline__ v16bf load_afrag(const float* __restrict__ row,
                                                   int kbase, int g, float s) {
  const float* p0 = row + kbase + g * 16;
  const float* p1 = p0 + 8;
  float4 a = *(const float4*)(p0);
  float4 b = *(const float4*)(p0 + 4);
  float4 c = *(const float4*)(p1);
  float4 d = *(const float4*)(p1 + 4);
  v16bf f;
  f[0]  = (bf16_t)(a.x * s); f[1]  = (bf16_t)(a.y * s);
  f[2]  = (bf16_t)(a.z * s); f[3]  = (bf16_t)(a.w * s);
  f[4]  = (bf16_t)(b.x * s); f[5]  = (bf16_t)(b.y * s);
  f[6]  = (bf16_t)(b.z * s); f[7]  = (bf16_t)(b.w * s);
  f[8]  = (bf16_t)(c.x * s); f[9]  = (bf16_t)(c.y * s);
  f[10] = (bf16_t)(c.z * s); f[11] = (bf16_t)(c.w * s);
  f[12] = (bf16_t)(d.x * s); f[13] = (bf16_t)(d.y * s);
  f[14] = (bf16_t)(d.z * s); f[15] = (bf16_t)(d.w * s);
  return f;
}

// bf16 source: the two 16B chunks are exactly the low/high VGPR halves of the
// fragment -> union bit-cast, zero shuffle VALU.
static __device__ __forceinline__ v16bf load_afrag(const bf16_t* __restrict__ row,
                                                   int kbase, int g, float /*s*/) {
  union { v16bf f; struct { v8bf lo, hi; } p; } u;
  u.p.lo = *(const v8bf*)(row + kbase + g * 16);
  u.p.hi = *(const v8bf*)(row + kbase + g * 16 + 8);
  return u.f;
}

// ---------------- degree / inverse degree ----------------
__global__ void deg_kernel(const int* __restrict__ dst, float* __restrict__ deg, int nE) {
  int e = blockIdx.x * blockDim.x + threadIdx.x;
  if (e < nE) atomicAdd(deg + dst[e], 1.0f);
}

__global__ void inv_kernel(float* __restrict__ deg, int n) {
  int i = blockIdx.x * blockDim.x + threadIdx.x;
  if (i < n) deg[i] = 1.0f / fmaxf(deg[i], 1.0f);
}

// ---------------- edge scatter-add aggregation (fp32 features, layer 1) ------
template <int K>
__global__ void edge_agg_f32(const float* __restrict__ feat, const int* __restrict__ src,
                             const int* __restrict__ dst, float* __restrict__ agg, int nE) {
  constexpr int C = K / 4;
  int gid = blockIdx.x * blockDim.x + threadIdx.x;
  int e = gid / C;
  if (e >= nE) return;
  int c = gid % C;
  int sN = src[e], dN = dst[e];
  float4 v = ((const float4*)(feat + (size_t)sN * K))[c];
  float* o = agg + (size_t)dN * K + c * 4;
  atomicAdd(o + 0, v.x);
  atomicAdd(o + 1, v.y);
  atomicAdd(o + 2, v.z);
  atomicAdd(o + 3, v.w);
}

// ---------------- edge scatter-add aggregation (bf16 features, layers 2/3) ---
// Gather reads are bf16 (half the bytes); accumulation stays fp32 for accuracy.
template <int K>
__global__ void edge_agg_bf16(const bf16_t* __restrict__ feat, const int* __restrict__ src,
                              const int* __restrict__ dst, float* __restrict__ agg, int nE) {
  constexpr int C = K / 8;
  int gid = blockIdx.x * blockDim.x + threadIdx.x;
  int e = gid / C;
  if (e >= nE) return;
  int c = gid % C;
  int sN = src[e], dN = dst[e];
  v8bf v = *(const v8bf*)(feat + (size_t)sN * K + c * 8);
  float* o = agg + (size_t)dN * K + c * 8;
#pragma unroll
  for (int i = 0; i < 8; ++i) atomicAdd(o + i, (float)v[i]);
}

// ---------------- fused SAGE GEMM: h = BN(relu?)(agg/deg @ Wl + bl + x @ Wr) --
template <typename FT, int K, bool BNRELU>
__global__ __launch_bounds__(128) void sage_gemm(
    const FT* __restrict__ x, const float* __restrict__ agg,
    const float* __restrict__ inv_deg,
    const float* __restrict__ Wl, const float* __restrict__ bl,
    const float* __restrict__ Wr,
    const float* __restrict__ gam, const float* __restrict__ bet,
    const float* __restrict__ rm,  const float* __restrict__ rv,
    bf16_t* __restrict__ out, int n_nodes) {
  constexpr int KT = K / 32;
  // Weights staged transposed [col][k] in bf16 so a B fragment (col = n,
  // K = kbase + g*16 + [0..15]) is 16 contiguous halves -> one 32B LDS read.
  __shared__ __align__(32) bf16_t WtL[64 * K];
  __shared__ __align__(32) bf16_t WtR[64 * K];
  const int colbase = blockIdx.y * 64;

  for (int idx = threadIdx.x; idx < 64 * K; idx += blockDim.x) {
    int k = idx / 64, col = idx % 64;                  // coalesced global W reads
    WtL[col * K + k] = (bf16_t)Wl[k * DHID + colbase + col];
    WtR[col * K + k] = (bf16_t)Wr[k * DHID + colbase + col];
  }
  __syncthreads();

  const int wave = threadIdx.x >> 5;
  const int lane = threadIdx.x & 31;
  const int g = lane >> 4;        // half-wave group
  const int m = lane & 15;        // A row / B-C column index
  const int ntiles = n_nodes >> 4;

  for (int tile = blockIdx.x * 4 + wave; tile < ntiles; tile += gridDim.x * 4) {
    const int row = tile * 16 + m;
    const FT* xrow = x + (size_t)row * K;
    const float* arow = agg + (size_t)row * K;
    const float s = inv_deg[row];

    v16bf ax[KT], aa[KT];
#pragma unroll
    for (int kt = 0; kt < KT; ++kt) {
      ax[kt] = load_afrag(xrow, kt * 32, g, 1.0f);
      aa[kt] = load_afrag(arow, kt * 32, g, s);       // fold 1/deg into the fragment
    }

    v8f acc[4] = {};
#pragma unroll
    for (int nt = 0; nt < 4; ++nt) {
      const size_t boff = (size_t)(nt * 16 + m) * K + g * 16;
#pragma unroll
      for (int kt = 0; kt < KT; ++kt) {
        v16bf bfl = *(const v16bf*)(WtL + boff + kt * 32);
        v16bf bfr = *(const v16bf*)(WtR + boff + kt * 32);
        acc[nt] = __builtin_amdgcn_wmma_f32_16x16x32_bf16(
            false, aa[kt], false, bfl, (short)0, acc[nt], false, false);
        acc[nt] = __builtin_amdgcn_wmma_f32_16x16x32_bf16(
            false, ax[kt], false, bfr, (short)0, acc[nt], false, false);
      }
    }

#pragma unroll
    for (int nt = 0; nt < 4; ++nt) {
      const int col = colbase + nt * 16 + m;
      const float bias = bl[col];
      float scale = 1.0f, shift = 0.0f;
      if (BNRELU) {
        float rs = rsqrtf(rv[col] + 1e-5f);
        scale = gam[col] * rs;
        shift = bet[col] - rm[col] * scale;
      }
#pragma unroll
      for (int v = 0; v < 8; ++v) {
        int r = tile * 16 + v + 8 * g;                // ISA C layout: row = v + 8g
        float val = acc[nt][v] + bias;
        if (BNRELU) val = fmaxf(val * scale + shift, 0.0f);
        out[(size_t)r * DHID + col] = (bf16_t)val;
      }
    }
  }
}

// ---------------- segment max pooling (ordered-uint encoding) ----------------
static __device__ __forceinline__ unsigned enc_f(float f) {
  unsigned u = __float_as_uint(f);
  return (u & 0x80000000u) ? ~u : (u | 0x80000000u);
}
static __device__ __forceinline__ float dec_f(unsigned u) {
  return (u & 0x80000000u) ? __uint_as_float(u & 0x7fffffffu) : __uint_as_float(~u);
}

__global__ void pool_init(unsigned* __restrict__ p, int n) {
  int i = blockIdx.x * blockDim.x + threadIdx.x;
  if (i < n) p[i] = enc_f(-3.402823466e38f);
}

__global__ void pool_max(const bf16_t* __restrict__ h, const int* __restrict__ batch,
                         unsigned* __restrict__ p, int nNodes) {
  int gid = blockIdx.x * blockDim.x + threadIdx.x;
  int node = gid / (DHID / 8);
  if (node >= nNodes) return;
  int c = gid % (DHID / 8);
  int b = batch[node];
  v8bf v = *(const v8bf*)(h + (size_t)node * DHID + c * 8);
  unsigned* o = p + (size_t)b * DHID + c * 8;
#pragma unroll
  for (int i = 0; i < 8; ++i) atomicMax(o + i, enc_f((float)v[i]));
}

__global__ void fc_kernel(const unsigned* __restrict__ pooled, const float* __restrict__ fcW,
                          const float* __restrict__ fcb, float* __restrict__ out, int G) {
  int gph = blockIdx.x * (blockDim.x / 32) + (threadIdx.x >> 5);
  int lane = threadIdx.x & 31;
  if (gph >= G) return;
  float sum = 0.0f;
  for (int d = lane; d < DHID; d += 32)
    sum += dec_f(pooled[(size_t)gph * DHID + d]) * fcW[d];
  for (int o = 16; o; o >>= 1) sum += __shfl_xor(sum, o, 32);
  if (lane == 0) out[gph] = sum + fcb[0];
}

// ---------------- host launch ----------------
static inline int cdiv(long long a, long long b) { return (int)((a + b - 1) / b); }
static inline size_t alup(size_t x) { return (x + 255) & ~(size_t)255; }

extern "C" void kernel_launch(void* const* d_in, const int* in_sizes, int n_in,
                              void* d_out, int out_size, void* d_ws, size_t ws_size,
                              hipStream_t stream) {
  const float* x     = (const float*)d_in[0];
  const int*   ei    = (const int*)d_in[1];
  const int*   batch = (const int*)d_in[2];
  const float* W1l = (const float*)d_in[3];
  const float* b1l = (const float*)d_in[4];
  const float* W1r = (const float*)d_in[5];
  const float* W2l = (const float*)d_in[6];
  const float* b2l = (const float*)d_in[7];
  const float* W2r = (const float*)d_in[8];
  const float* W3l = (const float*)d_in[9];
  const float* b3l = (const float*)d_in[10];
  const float* W3r = (const float*)d_in[11];
  const float* g1  = (const float*)d_in[12];
  const float* be1 = (const float*)d_in[13];
  const float* rm1 = (const float*)d_in[14];
  const float* rv1 = (const float*)d_in[15];
  const float* g2  = (const float*)d_in[16];
  const float* be2 = (const float*)d_in[17];
  const float* rm2 = (const float*)d_in[18];
  const float* rv2 = (const float*)d_in[19];
  const float* fcW = (const float*)d_in[20];
  const float* fcb = (const float*)d_in[21];

  const int N = in_sizes[2];
  const int E = in_sizes[1] / 2;
  const int G = out_size;
  const int* src = ei;
  const int* dst = ei + E;

  char* ws = (char*)d_ws;
  float*   invd = (float*)ws;   ws += alup((size_t)N * 4);
  float*   agg  = (float*)ws;   ws += alup((size_t)N * DHID * 4);   // fp32 accumulator
  bf16_t*  h1   = (bf16_t*)ws;  ws += alup((size_t)N * DHID * 2);   // bf16 activations
  bf16_t*  h2   = (bf16_t*)ws;  ws += alup((size_t)N * DHID * 2);
  unsigned* pooled = (unsigned*)ws;

  const dim3 gemmGrid(512, 2), gemmBlk(128);

  // degrees -> inverse degrees
  hipMemsetAsync(invd, 0, (size_t)N * 4, stream);
  deg_kernel<<<cdiv(E, 256), 256, 0, stream>>>(dst, invd, E);
  inv_kernel<<<cdiv(N, 256), 256, 0, stream>>>(invd, N);

  // layer 1 (K=64, fp32 input), BN+ReLU -> h1 (bf16)
  hipMemsetAsync(agg, 0, (size_t)N * NFEAT * 4, stream);
  edge_agg_f32<NFEAT><<<cdiv((long long)E * (NFEAT / 4), 256), 256, 0, stream>>>(
      x, src, dst, agg, E);
  sage_gemm<float, NFEAT, true><<<gemmGrid, gemmBlk, 0, stream>>>(
      x, agg, invd, W1l, b1l, W1r, g1, be1, rm1, rv1, h1, N);

  // layer 2 (K=128, bf16 input), BN+ReLU -> h2 (bf16)
  hipMemsetAsync(agg, 0, (size_t)N * DHID * 4, stream);
  edge_agg_bf16<DHID><<<cdiv((long long)E * (DHID / 8), 256), 256, 0, stream>>>(
      h1, src, dst, agg, E);
  sage_gemm<bf16_t, DHID, true><<<gemmGrid, gemmBlk, 0, stream>>>(
      h1, agg, invd, W2l, b2l, W2r, g2, be2, rm2, rv2, h2, N);

  // layer 3 (K=128, bf16 input), no BN/ReLU -> reuse h1 as h3
  hipMemsetAsync(agg, 0, (size_t)N * DHID * 4, stream);
  edge_agg_bf16<DHID><<<cdiv((long long)E * (DHID / 8), 256), 256, 0, stream>>>(
      h2, src, dst, agg, E);
  sage_gemm<bf16_t, DHID, false><<<gemmGrid, gemmBlk, 0, stream>>>(
      h2, agg, invd, W3l, b3l, W3r, nullptr, nullptr, nullptr, nullptr, h1, N);

  // global max pool + FC
  pool_init<<<cdiv((long long)G * DHID, 256), 256, 0, stream>>>(pooled, G * DHID);
  pool_max<<<cdiv((long long)N * (DHID / 8), 256), 256, 0, stream>>>(h1, batch, pooled, N);
  fc_kernel<<<cdiv(G, 8), 256, 0, stream>>>(pooled, fcW, fcb, (float*)d_out, G);
}
